// BatchedDTW_33543694581842
// MI455X (gfx1250) — matrix operand order repaired and checked by reading, not compiled, not run on hardware
//
#include <hip/hip_runtime.h>

typedef __attribute__((ext_vector_type(16))) _Float16 v16h;
typedef __attribute__((ext_vector_type(8)))  float    v8f;

#define T_LEN      512
#define N_NODE     64
#define C_DIM      32
#define ROW_STRIDE (N_NODE * C_DIM)   // 2048 floats between consecutive t
#define WIN        16
#define DSTRIDE    52                 // 48 band cols padded (bank-conflict free)

// Raw f32 tile data for one 16-row chunk, double-buffered for the pipeline.
struct RawTiles {
    float4 a[4];      // A frag raw: row[16h..16h+15]
    float4 b[3][4];   // B frag raw per band col-tile: yrow[16h..16h+15]
};

__device__ __forceinline__ void issue_chunk_loads(const float* __restrict__ xm,
                                                  const float* __restrict__ ym,
                                                  int ti, int h, int q16,
                                                  RawTiles& r) {
    const float* rowp = xm + (size_t)(16 * ti + q16) * ROW_STRIDE;
    const float4* ra  = (const float4*)(rowp + 16 * h);
    r.a[0] = ra[0]; r.a[1] = ra[1];
    const float4* ra2 = (const float4*)(rowp + 16 * h + 8);
    r.a[2] = ra2[0]; r.a[3] = ra2[1];
#pragma unroll
    for (int tjn = 0; tjn < 3; ++tjn) {
        int yrow = 16 * (ti - 1 + tjn) + q16;
        int yc   = yrow < 0 ? 0 : (yrow > T_LEN - 1 ? T_LEN - 1 : yrow);
        const float4* yp = (const float4*)(ym + (size_t)yc * ROW_STRIDE + 16 * h);
#pragma unroll
        for (int q = 0; q < 4; ++q) r.b[tjn][q] = yp[q];
    }
}

__device__ __forceinline__ v16h cvt_frag(const float4* v) {
    const float* f = (const float*)v;
    v16h o;
#pragma unroll
    for (int e = 0; e < 16; ++e) o[e] = (_Float16)f[e];
    return o;
}

// One chunk: 3x v_wmma_f32_16x16x32_f16 band tiles -> D in LDS -> D prefetched
// to registers -> 16 sequential DTW steps (pure VALU + shfl). Issues the NEXT
// chunk's global loads before the WMMA/DTW work so their latency is hidden.
__device__ __forceinline__ void process_chunk(
        int ti, int lane, int h, int q16,
        const float* __restrict__ xm, const float* __restrict__ ym,
        const float* x2s, const float* y2s, float* Dc,
        RawTiles& cur, RawTiles& nxt, float& r, float& top, float INF) {
    v16h af  = cvt_frag(cur.a);
    v16h bf0 = cvt_frag(cur.b[0]);
    v16h bf1 = cvt_frag(cur.b[1]);
    v16h bf2 = cvt_frag(cur.b[2]);

    // Prefetch next chunk (clamped re-load of the last chunk on the final iter;
    // avoids OOB global reads, result unused).
    const int tin = (ti + 1 > 31) ? 31 : ti + 1;
    issue_chunk_loads(xm, ym, tin, h, q16, nxt);

#pragma unroll
    for (int tjn = 0; tjn < 3; ++tjn) {
        v8f acc = {};
        acc = __builtin_amdgcn_wmma_f32_16x16x32_f16(
            false, af, false, (tjn == 0 ? bf0 : (tjn == 1 ? bf1 : bf2)),
            (short)0, acc, false, false);
#pragma unroll
        for (int rr = 0; rr < 8; ++rr) {
            const int M      = rr + 8 * h;
            const int a_glob = 16 * ti + M;
            const int c_glob = 16 * (ti - 1 + tjn) + q16;
            const int cc     = c_glob < 0 ? 0 : (c_glob > T_LEN - 1 ? T_LEN - 1 : c_glob);
            float d2 = x2s[a_glob] + y2s[cc] - 2.0f * acc[rr];
            float dv = sqrtf(fmaxf(d2, 0.0f));
            if (dv != dv)          dv = 0.0f;     // nan -> 0
            else if (dv > 3.0e38f) dv = 1000.0f;  // +inf -> 1000
            Dc[M * DSTRIDE + 16 * tjn + q16] = dv;
        }
    }

    // Pull this chunk's D values out of LDS up front (wave-ordered LDS makes
    // store->load within one wave safe without a barrier); the DTW chain then
    // has no LDS latency inside the dependent recurrence.
    float Dl[16], Dt[16];
#pragma unroll
    for (int mi = 0; mi < 16; ++mi) {
        Dl[mi] = Dc[mi * DSTRIDE + mi + lane];   // D[i-1, i-17+lane]
        Dt[mi] = Dc[mi * DSTRIDE + mi + 32];     // D[i-1, i+15]
    }

#pragma unroll
    for (int mi = 0; mi < 16; ++mi) {
        const int i = 16 * ti + mi + 1;
        float rn = __shfl_down(r, 1, 32);        // prev[j] = r_{l+1}
        if (lane == 31) rn = top;
        const int j = i - WIN + lane;
        const float nv = (j >= 1 && j <= T_LEN) ? (Dl[mi] + fminf(r, rn)) : INF;
        const float nt = (i + WIN <= T_LEN)     ? (Dt[mi] + top)          : INF;
        r = nv; top = nt;
    }
}

__global__ __launch_bounds__(32)
void dtw_band_kernel(const float* __restrict__ X, const float* __restrict__ Y,
                     float* __restrict__ dist) {
    __shared__ float x2s[T_LEN];
    __shared__ float y2s[T_LEN];
    __shared__ float Dc[16 * DSTRIDE];

    const int m    = blockIdx.x;          // m = b*64 + n
    const int b    = m >> 6;
    const int n    = m & 63;
    const int lane = threadIdx.x;
    const float INF = __builtin_inff();

    const size_t base = (size_t)b * (T_LEN * ROW_STRIDE) + (size_t)n * C_DIM;
    const float* xm = X + base;
    const float* ym = Y + base;

    // ---- squared norms (f32, exact) ----
    for (int t = lane; t < T_LEN; t += 32) {
        const float4* xr = (const float4*)(xm + (size_t)t * ROW_STRIDE);
        const float4* yr = (const float4*)(ym + (size_t)t * ROW_STRIDE);
        float sx = 0.f, sy = 0.f;
#pragma unroll
        for (int q = 0; q < 8; ++q) {
            float4 a = xr[q], c = yr[q];
            sx += a.x*a.x + a.y*a.y + a.z*a.z + a.w*a.w;
            sy += c.x*c.x + c.y*c.y + c.z*c.z + c.w*c.w;
        }
        x2s[t] = sx; y2s[t] = sy;
    }
    __syncthreads();

    const int h   = lane >> 4;
    const int q16 = lane & 15;

    // Band registers: lane l holds R[i-1, (i-1)-16+l]; 'top' = 33rd element.
    float r   = (lane == 16) ? 0.0f : INF;   // R[0,0] = 0
    float top = INF;

    RawTiles raw0, raw1;
    issue_chunk_loads(xm, ym, 0, h, q16, raw0);

    // Manual 2x unroll so the double-buffer selection is static (stays in VGPRs).
    for (int tt = 0; tt < 16; ++tt) {
        process_chunk(2 * tt,     lane, h, q16, xm, ym, x2s, y2s, Dc,
                      raw0, raw1, r, top, INF);
        process_chunk(2 * tt + 1, lane, h, q16, xm, ym, x2s, y2s, Dc,
                      raw1, raw0, r, top, INF);
    }

    // R[512,512] lives at lane j - i + 16 = 16
    const float res = __shfl(r, 16, 32);
    if (lane == 0) dist[m] = res;
}

// per_n = nan_to_num(mean over b), answer = mean over n
__global__ __launch_bounds__(64)
void dtw_reduce_kernel(const float* __restrict__ dist, float* __restrict__ out) {
    __shared__ float acc[64];
    const int nidx = threadIdx.x;
    float s = 0.f;
#pragma unroll
    for (int bb = 0; bb < 4; ++bb) s += dist[bb * 64 + nidx];
    s *= 0.25f;
    if (s != s)                           s = 0.f;     // nan -> 0
    else if (s > 3.0e38f || s < -3.0e38f) s = 1000.f;  // +/-inf -> 1000
    acc[nidx] = s;
    __syncthreads();
    for (int off = 32; off > 0; off >>= 1) {
        if (nidx < off) acc[nidx] += acc[nidx + off];
        __syncthreads();
    }
    if (nidx == 0) out[0] = acc[0] * (1.0f / 64.0f);
}

extern "C" void kernel_launch(void* const* d_in, const int* in_sizes, int n_in,
                              void* d_out, int out_size, void* d_ws, size_t ws_size,
                              hipStream_t stream) {
    const float* X = (const float*)d_in[0];   // (4,512,64,32) f32
    const float* Y = (const float*)d_in[1];   // (4,512,64,32) f32
    // d_in[2] = window (int scalar, ==16); kernel is specialized for w=16.
    float* dist = (float*)d_ws;               // 256 floats of scratch
    dtw_band_kernel<<<256, 32, 0, stream>>>(X, Y, dist);
    dtw_reduce_kernel<<<1, 64, 0, stream>>>(dist, (float*)d_out);
}
